// DecoderRNN_53910429499819
// MI455X (gfx1250) — matrix operand order repaired and checked by reading, not compiled, not run on hardware
//
#include <hip/hip_runtime.h>
#include <hip/hip_bf16.h>

// ---------------- constants from the reference ----------------
#define Bq   128
#define Pp   196
#define Ee   2048
#define Dd   512
#define Aa   512
#define Mm   512
#define Vv   10000
#define Ll   21
#define Tt   20          // L-1 decode steps

typedef unsigned short u16;
typedef unsigned int   u32;

typedef __bf16  bf16x16 __attribute__((ext_vector_type(16)));
typedef float   f32x8   __attribute__((ext_vector_type(8)));

union Frag16 { uint4 q[2]; u16 u[16]; bf16x16 v; };

static __device__ __forceinline__ u16 f2bf(float f) {
    u32 u = __float_as_uint(f);
    u = (u + 0x7FFFu + ((u >> 16) & 1u)) >> 16;   // round-to-nearest-even
    return (u16)u;
}
static __device__ __forceinline__ float bf2f(u16 h) {
    return __uint_as_float(((u32)h) << 16);
}
static __device__ __forceinline__ float sigf(float x) { return 1.0f / (1.0f + __expf(-x)); }

// ---- CDNA5 async global->LDS copy (ASYNCcnt-tracked), CDNA5 ISA ----
// Generic pointers to __shared__ carry the LDS offset in their low 32 bits on amdgcn.
static __device__ __forceinline__ void async_ld_b128(const void* g, const u16* ldsPtr) {
    u32 l = (u32)(unsigned long long)ldsPtr;
    asm volatile("global_load_async_to_lds_b128 %0, %1, off" :: "v"(l), "v"(g) : "memory");
}
static __device__ __forceinline__ void wait_async0() {
    asm volatile("s_wait_asynccnt 0x0" ::: "memory");
}

// ---------------- 0) stable descending argsort (B=128) ----------------
__global__ void order_kernel(const int* __restrict__ lens, int* __restrict__ order,
                             int* __restrict__ dec_len) {
    __shared__ int sl[Bq];
    int i = threadIdx.x;
    sl[i] = lens[i];
    __syncthreads();
    int li = sl[i], r = 0;
    for (int j = 0; j < Bq; ++j) {
        int lj = sl[j];
        r += (lj > li) || (lj == li && j < i);
    }
    order[r]   = i;
    dec_len[r] = li - 1;
}

// ---------------- 1) gather encoder_out[order] -> bf16 ----------------
__global__ void gather_enc_kernel(const float* __restrict__ enc, const int* __restrict__ order,
                                  u16* __restrict__ out) {
    int bp = blockIdx.x;                       // [0, B*P)
    int b  = bp / Pp, p = bp % Pp;
    int e  = blockIdx.y * 256 + threadIdx.x;   // [0, E)
    out[(size_t)bp * Ee + e] =
        f2bf(enc[((size_t)order[b] * Pp + p) * Ee + e]);
}

// ---------------- 2) weight conversion kernels (once per call) ----------------
__global__ void bf16_copy_kernel(u16* __restrict__ dst, const float* __restrict__ src, int n) {
    int i = blockIdx.x * 256 + threadIdx.x;
    if (i < n) dst[i] = f2bf(src[i]);
}
// [2048,1024] = [init_h_w | init_c_w]
__global__ void cat_init_w_kernel(u16* __restrict__ dst, const float* __restrict__ hw,
                                  const float* __restrict__ cw) {
    int i = blockIdx.x * 256 + threadIdx.x;    // n = 2048*1024
    if (i >= Ee * 2 * Dd) return;
    int k = i >> 10, c = i & 1023;
    float v = (c < Dd) ? hw[k * Dd + c] : cw[k * Dd + (c - Dd)];
    dst[i] = f2bf(v);
}
__global__ void cat_init_b_kernel(float* __restrict__ dst, const float* __restrict__ hb,
                                  const float* __restrict__ cb) {
    int i = blockIdx.x * 256 + threadIdx.x;
    if (i < Dd)            dst[i] = hb[i];
    else if (i < 2 * Dd)   dst[i] = cb[i - Dd];
}
// W_lstm [3072,2048]: rows 0..2559 = w_ih^T, rows 2560..3071 = w_hh^T
__global__ void lstm_w_kernel(u16* __restrict__ dst, const float* __restrict__ wih,
                              const float* __restrict__ whh) {
    int i = blockIdx.x * 256 + threadIdx.x;    // n = 3072*2048
    if (i >= 3072 * 2048) return;
    int k = i >> 11, j = i & 2047;
    float v = (k < Mm + Ee) ? wih[(size_t)j * (Mm + Ee) + k]
                            : whh[(size_t)j * Dd + (k - (Mm + Ee))];
    dst[i] = f2bf(v);
}
__global__ void lstm_b_kernel(float* __restrict__ dst, const float* __restrict__ bih,
                              const float* __restrict__ bhh) {
    int i = blockIdx.x * 256 + threadIdx.x;
    if (i < 4 * Dd) dst[i] = bih[i] + bhh[i];
}
// W_fc [512,10000] = fc_w^T
__global__ void fc_t_kernel(u16* __restrict__ dst, const float* __restrict__ fcw) {
    int i = blockIdx.x * 256 + threadIdx.x;    // n = 512*10000
    if (i >= Dd * Vv) return;
    int k = i / Vv, n = i % Vv;
    dst[i] = f2bf(fcw[(size_t)n * Dd + k]);
}

// ---------------- 3) mean over P (from bf16 enc) ----------------
__global__ void mean_kernel(const u16* __restrict__ enc, float* __restrict__ mean) {
    int b = blockIdx.y;
    int e = blockIdx.x * 256 + threadIdx.x;
    const u16* base = enc + (size_t)b * Pp * Ee + e;
    float s = 0.f;
    for (int p = 0; p < Pp; ++p) s += bf2f(base[(size_t)p * Ee]);
    mean[b * Ee + e] = s * (1.0f / (float)Pp);
}

// ---------------- generic WMMA bf16 GEMM ----------------
// C[M,N] = A[M,K] * B[K,N] + bias.  A: float (converted on the fly) or bf16.
// Block tile 64x64x64, double-buffered LDS, B (and bf16 A) staged with
// GLOBAL_LOAD_ASYNC_TO_LDS_B128; 256 threads = 8 waves, wave tile 16x32.
// Requires: M % 64 == 0, K % 64 == 0.  B rows may be read up to 64 cols past N
// (caller pads); C columns >= N are never stored.
template <typename AT, int OUT_BF16>
__global__ __launch_bounds__(256)
void gemm_wmma_kernel(const AT* __restrict__ A, const u16* __restrict__ Bm,
                      const float* __restrict__ bias, void* __restrict__ Cout,
                      int M, int N, int K) {
    __shared__ u16 As[2][64 * 72];
    __shared__ u16 Bs[2][64 * 72];

    const int tid  = threadIdx.x;
    const int lane = tid & 31;
    const int w    = tid >> 5;
    const int wm   = w & 3;          // 4 m-tiles of 16
    const int wn   = w >> 2;         // 2 n-quadrants of 32
    const int lm   = lane & 15;
    const int lh   = lane >> 4;
    const int m0   = blockIdx.y * 64;
    const int n0   = blockIdx.x * 64;

    const int r0 = tid >> 3;          // 0..31 (two chunks: +0, +32)
    const int c0 = (tid & 7) * 8;     // 0..56

    f32x8 acc[2];
    acc[0] = (f32x8)0.0f;
    acc[1] = (f32x8)0.0f;

    const int NT = K >> 6;
    float aregs[16];

    // ---- prologue: stage tile 0 into buffer 0 ----
    {
#pragma unroll
        for (int i = 0; i < 2; ++i) {
            int row = r0 + i * 32;
            async_ld_b128(Bm + (size_t)row * N + n0 + c0, &Bs[0][row * 72 + c0]);
        }
        if constexpr (sizeof(AT) == 2) {
#pragma unroll
            for (int i = 0; i < 2; ++i) {
                int row = r0 + i * 32;
                async_ld_b128((const u16*)A + (size_t)(m0 + row) * K + c0,
                              &As[0][row * 72 + c0]);
            }
        } else {
#pragma unroll
            for (int i = 0; i < 2; ++i) {
                int row = r0 + i * 32;
                const float* src = (const float*)A + (size_t)(m0 + row) * K + c0;
                *(float4*)&aregs[i * 8]     = *(const float4*)src;
                *(float4*)&aregs[i * 8 + 4] = *(const float4*)(src + 4);
            }
#pragma unroll
            for (int i = 0; i < 2; ++i) {
                int row = r0 + i * 32;
                u16 tmp[8];
#pragma unroll
                for (int j = 0; j < 8; ++j) tmp[j] = f2bf(aregs[i * 8 + j]);
                *(uint4*)&As[0][row * 72 + c0] = *(const uint4*)tmp;
            }
        }
    }

    int cur = 0;
    for (int kt = 0; kt < NT; ++kt) {
        wait_async0();
        __syncthreads();
        const int nxt  = cur ^ 1;
        const bool more = (kt + 1 < NT);

        // ---- issue next tile while current computes ----
        if (more) {
            const int k0 = (kt + 1) << 6;
#pragma unroll
            for (int i = 0; i < 2; ++i) {
                int row = r0 + i * 32;
                async_ld_b128(Bm + (size_t)(k0 + row) * N + n0 + c0,
                              &Bs[nxt][row * 72 + c0]);
            }
            if constexpr (sizeof(AT) == 2) {
#pragma unroll
                for (int i = 0; i < 2; ++i) {
                    int row = r0 + i * 32;
                    async_ld_b128((const u16*)A + (size_t)(m0 + row) * K + k0 + c0,
                                  &As[nxt][row * 72 + c0]);
                }
            } else {
#pragma unroll
                for (int i = 0; i < 2; ++i) {
                    int row = r0 + i * 32;
                    const float* src = (const float*)A + (size_t)(m0 + row) * K + k0 + c0;
                    *(float4*)&aregs[i * 8]     = *(const float4*)src;
                    *(float4*)&aregs[i * 8 + 4] = *(const float4*)(src + 4);
                }
            }
        }

        // ---- compute current tile: 2 K-substeps of 32 ----
#pragma unroll
        for (int ks = 0; ks < 2; ++ks) {
            Frag16 af, bf0, bf1;
            const int arow = wm * 16 + lm;
            const int kb   = ks * 32 + lh * 8;   // CDNA5 A 16x32 bf16 lane layout
            af.q[0] = *(const uint4*)&As[cur][arow * 72 + kb];
            af.q[1] = *(const uint4*)&As[cur][arow * 72 + kb + 16];
            const int brow = ks * 32 + lh * 8;   // B: lane = N column, 16 K per lane (same K order as A)
            const int coln = wn * 32 + lm;
#pragma unroll
            for (int s = 0; s < 16; ++s) {
                const int kk = brow + (s & 7) + (s >> 3) * 16;
                bf0.u[s] = Bs[cur][kk * 72 + coln];
                bf1.u[s] = Bs[cur][kk * 72 + coln + 16];
            }
            acc[0] = __builtin_amdgcn_wmma_f32_16x16x32_bf16(
                false, af.v, false, bf0.v, (short)0, acc[0], false, false);
            acc[1] = __builtin_amdgcn_wmma_f32_16x16x32_bf16(
                false, af.v, false, bf1.v, (short)0, acc[1], false, false);
        }

        // ---- store prefetched fp32 A regs (converted) into next buffer ----
        if (more) {
            if constexpr (sizeof(AT) == 4) {
#pragma unroll
                for (int i = 0; i < 2; ++i) {
                    int row = r0 + i * 32;
                    u16 tmp[8];
#pragma unroll
                    for (int j = 0; j < 8; ++j) tmp[j] = f2bf(aregs[i * 8 + j]);
                    *(uint4*)&As[nxt][row * 72 + c0] = *(const uint4*)tmp;
                }
            }
        }
        cur = nxt;
    }

    // ---- epilogue: C layout (VGPR e: M = e + 8*lh, N = lm) ----
#pragma unroll
    for (int sn = 0; sn < 2; ++sn) {
        int n = n0 + wn * 32 + sn * 16 + lm;
        if (n >= N) continue;
        float bv = bias ? bias[n] : 0.0f;
#pragma unroll
        for (int e = 0; e < 8; ++e) {
            int m = m0 + wm * 16 + 8 * lh + e;
            float val = acc[sn][e] + bv;
            if (OUT_BF16) ((u16*)Cout)[(size_t)m * N + n] = f2bf(val);
            else          ((float*)Cout)[(size_t)m * N + n] = val;
        }
    }
}

// ---------------- split h0/c0 ----------------
__global__ void split_hc_kernel(const float* __restrict__ hc, float* __restrict__ h,
                                float* __restrict__ c) {
    int i = blockIdx.x * 256 + threadIdx.x;     // 128*512
    int b = i >> 9, j = i & 511;
    h[i] = hc[b * 2 * Dd + j];
    c[i] = hc[b * 2 * Dd + Dd + j];
}

// ---------------- e[b,p] = relu(att1+att2) . full_att_w + b ----------------
__global__ __launch_bounds__(256)
void att_e_kernel(const u16* __restrict__ att1, const float* __restrict__ att2,
                  const float* __restrict__ faw, const float* __restrict__ fab,
                  float* __restrict__ e) {
    int row  = blockIdx.x * 8 + (threadIdx.x >> 5);   // [0, B*P)
    int lane = threadIdx.x & 31;
    int b    = row / Pp;
    const u16*   a1 = att1 + (size_t)row * Aa;
    const float* a2 = att2 + b * Aa;
    float s = 0.f;
#pragma unroll 4
    for (int j = lane; j < Aa; j += 32) {
        float v = bf2f(a1[j]) + a2[j];
        v = v > 0.f ? v : 0.f;
        s += v * faw[j];
    }
#pragma unroll
    for (int off = 16; off > 0; off >>= 1) s += __shfl_xor(s, off, 32);
    if (lane == 0) e[row] = s + fab[0];
}

// ---------------- softmax over P; write alpha + masked alphas out ----------------
__global__ void softmax_alpha_kernel(const float* __restrict__ e, float* __restrict__ alpha,
                                     float* __restrict__ out_alphas,
                                     const int* __restrict__ dec_len, int t) {
    int b = blockIdx.x, tid = threadIdx.x;
    __shared__ float sv[256];
    float v = (tid < Pp) ? e[b * Pp + tid] : -3.0e38f;
    sv[tid] = v; __syncthreads();
    for (int s = 128; s > 0; s >>= 1) { if (tid < s) sv[tid] = fmaxf(sv[tid], sv[tid + s]); __syncthreads(); }
    float mx = sv[0]; __syncthreads();
    float ex = (tid < Pp) ? __expf(v - mx) : 0.f;
    sv[tid] = ex; __syncthreads();
    for (int s = 128; s > 0; s >>= 1) { if (tid < s) sv[tid] += sv[tid + s]; __syncthreads(); }
    float inv = 1.f / sv[0];
    if (tid < Pp) {
        float a = ex * inv;
        alpha[b * Pp + tid] = a;
        int active = (t < dec_len[b]);
        out_alphas[((size_t)b * Tt + t) * Pp + tid] = active ? a : 0.f;
    }
}

// ---------------- awe[b,e] = sum_p alpha[b,p] * enc_bf16[b,p,e] ----------------
__global__ __launch_bounds__(256)
void awe_kernel(const u16* __restrict__ enc, const float* __restrict__ alpha,
                float* __restrict__ awe) {
    int b = blockIdx.y;
    int e = blockIdx.x * 256 + threadIdx.x;
    __shared__ float sal[Pp];
    if (threadIdx.x < Pp) sal[threadIdx.x] = alpha[b * Pp + threadIdx.x];
    __syncthreads();
    const u16* base = enc + (size_t)b * Pp * Ee + e;
    float s = 0.f;
#pragma unroll 4
    for (int p = 0; p < Pp; ++p) s += sal[p] * bf2f(base[(size_t)p * Ee]);
    awe[b * Ee + e] = s;
}

// ---------------- z = [emb(caps[:,t]) | sigmoid(gpre)*awe | h] ----------------
__global__ void z_kernel(float* __restrict__ z, const float* __restrict__ emb_table,
                         const int* __restrict__ caps, const int* __restrict__ order,
                         const float* __restrict__ gpre, const float* __restrict__ awe,
                         const float* __restrict__ h, int t) {
    int b = blockIdx.y;
    int i = blockIdx.x * 256 + threadIdx.x;    // [0, 3072)
    float v;
    if (i < Mm) {
        int cap = caps[order[b] * Ll + t];
        v = emb_table[(size_t)cap * Mm + i];
    } else if (i < Mm + Ee) {
        int j = i - Mm;
        v = sigf(gpre[b * Ee + j]) * awe[b * Ee + j];
    } else {
        v = h[b * Dd + (i - Mm - Ee)];
    }
    z[(size_t)b * (Mm + Ee + Dd) + i] = v;
}

// ---------------- LSTM cell elementwise ----------------
__global__ void lstm_cell_kernel(const float* __restrict__ g, float* __restrict__ h,
                                 float* __restrict__ c) {
    int b = blockIdx.y;
    int j = blockIdx.x * 256 + threadIdx.x;    // [0, 512)
    const float* gb = g + (size_t)b * 4 * Dd;
    float ig = sigf(gb[j]);
    float fg = sigf(gb[Dd + j]);
    float gg = tanhf(gb[2 * Dd + j]);
    float og = sigf(gb[3 * Dd + j]);
    float cn = fg * c[b * Dd + j] + ig * gg;
    c[b * Dd + j] = cn;
    h[b * Dd + j] = og * tanhf(cn);
}

// ---------------- masked copy of predictions into d_out ----------------
__global__ void pred_mask_kernel(const float* __restrict__ pred, float* __restrict__ out,
                                 const int* __restrict__ dec_len, int t) {
    int b = blockIdx.y;
    int v = blockIdx.x * 256 + threadIdx.x;
    if (v >= Vv) return;
    int active = (t < dec_len[b]);
    out[((size_t)b * Tt + t) * Vv + v] = active ? pred[(size_t)b * Vv + v] : 0.f;
}

// =====================================================================
extern "C" void kernel_launch(void* const* d_in, const int* in_sizes, int n_in,
                              void* d_out, int out_size, void* d_ws, size_t ws_size,
                              hipStream_t stream) {
    (void)in_sizes; (void)n_in; (void)out_size; (void)ws_size;
    const float* encoder_out = (const float*)d_in[0];
    const int*   caps        = (const int*)d_in[1];
    const int*   cap_len     = (const int*)d_in[2];
    const float* emb_table   = (const float*)d_in[3];
    const float* enc_att_w   = (const float*)d_in[4];
    const float* enc_att_b   = (const float*)d_in[5];
    const float* dec_att_w   = (const float*)d_in[6];
    const float* dec_att_b   = (const float*)d_in[7];
    const float* full_att_w  = (const float*)d_in[8];
    const float* full_att_b  = (const float*)d_in[9];
    const float* init_h_w    = (const float*)d_in[10];
    const float* init_h_b    = (const float*)d_in[11];
    const float* init_c_w    = (const float*)d_in[12];
    const float* init_c_b    = (const float*)d_in[13];
    const float* f_beta_w    = (const float*)d_in[14];
    const float* f_beta_b    = (const float*)d_in[15];
    const float* lstm_w_ih   = (const float*)d_in[16];
    const float* lstm_b_ih   = (const float*)d_in[17];
    const float* lstm_w_hh   = (const float*)d_in[18];
    const float* lstm_b_hh   = (const float*)d_in[19];
    const float* fc_w        = (const float*)d_in[20];
    const float* fc_b        = (const float*)d_in[21];

    float* out_pred   = (float*)d_out;                              // [B,T,V]
    float* out_alphas = (float*)d_out + (size_t)Bq * Tt * Vv;       // [B,T,P]

    // ---- workspace carve-up ----
    char*  ws  = (char*)d_ws;
    size_t off = 0;
    auto carve = [&](size_t bytes) -> char* {
        char* p = ws + off;
        off = (off + bytes + 255) & ~(size_t)255;
        return p;
    };
    int*   order   = (int*)  carve(Bq * 4);
    int*   dec_len = (int*)  carve(Bq * 4);
    u16*   enc_bf  = (u16*)  carve((size_t)Bq * Pp * Ee * 2);   // 102.8 MB
    u16*   att1    = (u16*)  carve((size_t)Bq * Pp * Aa * 2);   // 25.7 MB
    u16*   Wcat    = (u16*)  carve((size_t)Ee * 2 * Dd * 2);
    u16*   Wenc    = (u16*)  carve((size_t)Ee * Aa * 2);
    u16*   Wdec    = (u16*)  carve((size_t)Dd * Aa * 2);
    u16*   Wfb     = (u16*)  carve((size_t)Dd * Ee * 2);
    u16*   Wlstm   = (u16*)  carve((size_t)3072 * 2048 * 2);
    u16*   Wfc     = (u16*)  carve((size_t)Dd * Vv * 2 + 256);  // pad: edge-tile reads
    float* b_init  = (float*)carve(2 * Dd * 4);
    float* b_lstm  = (float*)carve(4 * Dd * 4);
    float* meanb   = (float*)carve((size_t)Bq * Ee * 4);
    float* hc      = (float*)carve((size_t)Bq * 2 * Dd * 4);
    float* hbuf    = (float*)carve((size_t)Bq * Dd * 4);
    float* cbuf    = (float*)carve((size_t)Bq * Dd * 4);
    float* att2    = (float*)carve((size_t)Bq * Aa * 4);
    float* ebuf    = (float*)carve((size_t)Bq * Pp * 4);
    float* alpha   = (float*)carve((size_t)Bq * Pp * 4);
    float* awe     = (float*)carve((size_t)Bq * Ee * 4);
    float* gpre    = (float*)carve((size_t)Bq * Ee * 4);
    float* zbuf    = (float*)carve((size_t)Bq * 3072 * 4);
    float* gbuf    = (float*)carve((size_t)Bq * 4 * Dd * 4);
    float* pred    = (float*)carve((size_t)Bq * Vv * 4);

    auto cdiv = [](int a, int b) { return (a + b - 1) / b; };

    // ---- one-time: sort, gather/convert, weight prep ----
    order_kernel<<<1, Bq, 0, stream>>>(cap_len, order, dec_len);
    gather_enc_kernel<<<dim3(Bq * Pp, Ee / 256), 256, 0, stream>>>(encoder_out, order, enc_bf);

    bf16_copy_kernel<<<cdiv(Ee * Aa, 256), 256, 0, stream>>>(Wenc, enc_att_w, Ee * Aa);
    bf16_copy_kernel<<<cdiv(Dd * Aa, 256), 256, 0, stream>>>(Wdec, dec_att_w, Dd * Aa);
    bf16_copy_kernel<<<cdiv(Dd * Ee, 256), 256, 0, stream>>>(Wfb, f_beta_w, Dd * Ee);
    cat_init_w_kernel<<<cdiv(Ee * 2 * Dd, 256), 256, 0, stream>>>(Wcat, init_h_w, init_c_w);
    cat_init_b_kernel<<<cdiv(2 * Dd, 256), 256, 0, stream>>>(b_init, init_h_b, init_c_b);
    lstm_w_kernel<<<cdiv(3072 * 2048, 256), 256, 0, stream>>>(Wlstm, lstm_w_ih, lstm_w_hh);
    lstm_b_kernel<<<cdiv(4 * Dd, 256), 256, 0, stream>>>(b_lstm, lstm_b_ih, lstm_b_hh);
    fc_t_kernel<<<cdiv(Dd * Vv, 256), 256, 0, stream>>>(Wfc, fc_w);

    mean_kernel<<<dim3(Ee / 256, Bq), 256, 0, stream>>>(enc_bf, meanb);

    // h0|c0 = mean @ [init_h_w|init_c_w] + [b_h|b_c]     [128,2048]x[2048,1024]
    gemm_wmma_kernel<float, 0><<<dim3((2 * Dd) / 64, Bq / 64), 256, 0, stream>>>(
        meanb, Wcat, b_init, hc, Bq, 2 * Dd, Ee);
    split_hc_kernel<<<cdiv(Bq * Dd, 256), 256, 0, stream>>>(hc, hbuf, cbuf);

    // att1 = enc @ enc_att_w + b   [25088,2048]x[2048,512] -> bf16
    gemm_wmma_kernel<u16, 1><<<dim3(Aa / 64, (Bq * Pp) / 64), 256, 0, stream>>>(
        enc_bf, Wenc, enc_att_b, att1, Bq * Pp, Aa, Ee);

    // ---- decode steps ----
    for (int t = 0; t < Tt; ++t) {
        // att2 = h @ dec_att_w + b      [128,512]x[512,512]
        gemm_wmma_kernel<float, 0><<<dim3(Aa / 64, Bq / 64), 256, 0, stream>>>(
            hbuf, Wdec, dec_att_b, att2, Bq, Aa, Dd);
        // e = relu(att1 + att2) . full_att_w + b
        att_e_kernel<<<(Bq * Pp) / 8, 256, 0, stream>>>(att1, att2, full_att_w, full_att_b, ebuf);
        // alpha = softmax_P(e); masked alphas -> d_out
        softmax_alpha_kernel<<<Bq, 256, 0, stream>>>(ebuf, alpha, out_alphas, dec_len, t);
        // awe = sum_p alpha * enc
        awe_kernel<<<dim3(Ee / 256, Bq), 256, 0, stream>>>(enc_bf, alpha, awe);
        // gpre = h @ f_beta_w + b       [128,512]x[512,2048]
        gemm_wmma_kernel<float, 0><<<dim3(Ee / 64, Bq / 64), 256, 0, stream>>>(
            hbuf, Wfb, f_beta_b, gpre, Bq, Ee, Dd);
        // z = [emb | sigmoid(gpre)*awe | h]
        z_kernel<<<dim3(3072 / 256, Bq), 256, 0, stream>>>(zbuf, emb_table, caps, order,
                                                           gpre, awe, hbuf, t);
        // g = z @ Wlstm + (b_ih+b_hh)   [128,3072]x[3072,2048]
        gemm_wmma_kernel<float, 0><<<dim3((4 * Dd) / 64, Bq / 64), 256, 0, stream>>>(
            zbuf, Wlstm, b_lstm, gbuf, Bq, 4 * Dd, 3072);
        // LSTM cell update (h, c)
        lstm_cell_kernel<<<dim3(Dd / 256, Bq), 256, 0, stream>>>(gbuf, hbuf, cbuf);
        // pred = h @ fc_w^T + fc_b      [128,512]x[512,10000]
        gemm_wmma_kernel<float, 0><<<dim3(cdiv(Vv, 64), Bq / 64), 256, 0, stream>>>(
            hbuf, Wfc, fc_b, pred, Bq, Vv, Dd);
        // masked write into d_out
        pred_mask_kernel<<<dim3(cdiv(Vv, 256), Bq), 256, 0, stream>>>(pred, out_pred, dec_len, t);
    }
}
